// FlashMLA_88536455840361
// MI455X (gfx1250) — matrix operand order repaired and checked by reading, not compiled, not run on hardware
//
#include <hip/hip_runtime.h>
#include <hip/hip_bf16.h>
#include <stdint.h>

// ---------------------------------------------------------------------------
// Causal FlashAttention forward, fp32 in/out, f16 WMMA compute (gfx1250).
// B=4, SQ=SK=2048, H=16, D=128.
//
// Design (wave32 / WMMA 16x16x32 f16):
//   * Each workgroup: 256 threads (8 waves), BM=128 query rows for one (b,h).
//   * Each wave owns 16 query rows. We compute S^T = K*Q^T and O^T = V^T*P^T
//     so softmax rows live one-per-lane (reductions = regs + shfl_xor(16)),
//     and P^T WMMA-B fragments are built register-only (no LDS round trip).
//   * K tiles prefetched with global_load_async_to_lds_b128 (double-buffered
//     fp32 stage, ASYNCcnt), converted to f16 tile in LDS. V loaded directly,
//     converted+transposed into LDS (f16). All fragments: ds_load_b128.
//   * Raw v_exp_f32 (inputs always <= 0; -inf -> 0 is exactly what we want).
//   * Output transposed through LDS for coalesced float4 global stores.
// ---------------------------------------------------------------------------

typedef __attribute__((ext_vector_type(16))) _Float16 v16h;
typedef __attribute__((ext_vector_type(8)))  _Float16 v8h;
typedef __attribute__((ext_vector_type(8)))  float    v8f;

constexpr int FA_B  = 4;
constexpr int FA_SQ = 2048;
constexpr int FA_SK = 2048;
constexpr int FA_H  = 16;
constexpr int FA_D  = 128;

constexpr int BM = 128;   // query rows per workgroup
constexpr int BN = 64;    // key rows per tile

constexpr int KST = 136;  // kt row stride in halves (128 + 8 pad, keeps 16B align)
constexpr int VST = 72;   // vt row stride in halves (64 + 8 pad, keeps 16B align)
constexpr int OST = 132;  // output staging row stride in floats (16B aligned)

// SCALE = 1/sqrt(128); folded with log2(e) so we can use exp2.
#define CEXP (0.08838834764831845f * 1.4426950408889634f)

// LDS layout (dynamic shared):
//   [0, 32768)        stageK buf0 (fp32)
//   [32768, 65536)    stageK buf1 (fp32)
//   [65536, 82944)    kt: f16 [64][KST]
//   [82944, 101376)   vt: f16 [128][VST]
//   epilogue reuses [0, 67584) as per-wave fp32 output staging.
constexpr int SM_KT    = 65536;
constexpr int SM_VT    = 82944;
constexpr int SM_TOTAL = 101376;

extern __shared__ __align__(16) unsigned char smem[];

__device__ __forceinline__ v16h load_frag16(const _Float16* p) {
    // A-matrix 16x32 f16 fragment for this lane: 16 contiguous halves
    // -> two ds_load_b128.
    v8h lo = *(const v8h*)(p);
    v8h hi = *(const v8h*)(p + 8);
    v16h f;
#pragma unroll
    for (int i = 0; i < 8; ++i) { f[i] = lo[i]; f[i + 8] = hi[i]; }
    return f;
}

__device__ __forceinline__ uint32_t pk_f16x2(float a, float b) {
    // v_cvt_pk_f16_f32
    union { _Float16 h[2]; uint32_t u; } p;
    p.h[0] = (_Float16)a;
    p.h[1] = (_Float16)b;
    return p.u;
}

__global__ __launch_bounds__(256, 1)
void fa_fwd_wmma(const float* __restrict__ Q, const float* __restrict__ K,
                 const float* __restrict__ V, float* __restrict__ O) {
    const int tid  = threadIdx.x;
    const int wave = tid >> 5;
    const int lane = tid & 31;
    const int g    = lane >> 4;   // half-wave group
    const int lm   = lane & 15;   // lane-in-group == query column m for C layout
    const int h    = blockIdx.y;
    const int b    = blockIdx.z;
    const int q0   = blockIdx.x * BM + wave * 16;   // first query row of this wave

    _Float16* kt = (_Float16*)(smem + SM_KT);
    _Float16* vt = (_Float16*)(smem + SM_VT);
    const uint32_t smBase = (uint32_t)(uintptr_t)(&smem[0]);  // LDS byte offset base

    // ---- Q fragments (B-matrix layout: lane=col m, halves = contiguous d) ----
    const float* qrow = Q + (((size_t)b * FA_SQ + q0 + lm) * FA_H + h) * FA_D;
    v16h qf[4];
#pragma unroll
    for (int c = 0; c < 4; ++c) {
        const float* p = qrow + 32 * c + 16 * g;
        v16h f;
#pragma unroll
        for (int j = 0; j < 16; j += 4) {
            float4 x = *(const float4*)(p + j);
            f[j + 0] = (_Float16)x.x; f[j + 1] = (_Float16)x.y;
            f[j + 2] = (_Float16)x.z; f[j + 3] = (_Float16)x.w;
        }
        qf[c] = f;
    }

    v8f oacc[8];
#pragma unroll
    for (int i = 0; i < 8; ++i) {
#pragma unroll
        for (int r = 0; r < 8; ++r) oacc[i][r] = 0.0f;
    }
    float m_run = -__builtin_inff();
    float l_run = 0.0f;

    const int nIters = (blockIdx.x + 1) * (BM / BN);  // causal: tiles up to block diag

    // ---- async K tile issue: each wave copies 8 rows, 512B per instruction ----
    auto issueK = [&](int kt0, int bufOff) {
        const size_t kbase = (((size_t)b * FA_SK + kt0) * FA_H + h) * FA_D;
#pragma unroll
        for (int rr = 0; rr < 8; ++rr) {
            const int row = wave * 8 + rr;
            uint32_t lds = smBase + (uint32_t)bufOff + (uint32_t)(row * 512 + lane * 16);
            uint64_t ga  = (uint64_t)(uintptr_t)(K + kbase + (size_t)row * (FA_H * FA_D) + lane * 4);
            asm volatile("global_load_async_to_lds_b128 %0, %1, off"
                         :: "v"(lds), "v"(ga) : "memory");
        }
    };

    issueK(0, 0);
    int buf = 0;

    for (int it = 0; it < nIters; ++it) {
        const int kt0 = it * BN;

        asm volatile("s_wait_asynccnt 0" ::: "memory");
        __syncthreads();  // stage[buf] complete everywhere; prior compute done

        if (it + 1 < nIters) issueK(kt0 + BN, (buf ^ 1) * 32768);

        // ---- convert staged K (fp32) -> kt (f16, row-major, padded) ----
        const float* stg = (const float*)(smem + buf * 32768);
#pragma unroll
        for (int j = 0; j < 8; ++j) {
            const int lin = (j * 256 + tid) * 4;
            const int n = lin >> 7, d = lin & 127;
            float4 x = *(const float4*)(stg + lin);
            union { uint32_t u[2]; unsigned long long ull; } pk;
            pk.u[0] = pk_f16x2(x.x, x.y);
            pk.u[1] = pk_f16x2(x.z, x.w);
            *(unsigned long long*)(kt + n * KST + d) = pk.ull;
        }

        // ---- load V (fp32, coalesced) -> vt transposed (f16) + prefetch next ----
        const float* vbase = V + (((size_t)b * FA_SK + kt0) * FA_H + h) * FA_D;
#pragma unroll
        for (int j = 0; j < 8; ++j) {
            const int lin = (j * 256 + tid) * 4;
            const int n = lin >> 7, d = lin & 127;
            const float* vp = vbase + (size_t)n * (FA_H * FA_D) + d;
            float4 x = *(const float4*)vp;
            vt[(d + 0) * VST + n] = (_Float16)x.x;
            vt[(d + 1) * VST + n] = (_Float16)x.y;
            vt[(d + 2) * VST + n] = (_Float16)x.z;
            vt[(d + 3) * VST + n] = (_Float16)x.w;
            if (it + 1 < nIters)
                __builtin_prefetch(vp + (size_t)BN * FA_H * FA_D, 0, 1);
        }
        __syncthreads();  // kt/vt visible to all waves

        // wave-uniform: is this tile at least partially unmasked for our rows?
        const bool active = (kt0 <= q0 + 15);
        if (active) {
            // ---- S^T tiles: D(16n x 16m) = A(K rows 16x32d) x B(Q^T 32d x 16m) ----
            v8f sacc[4];
#pragma unroll
            for (int t = 0; t < 4; ++t) {
#pragma unroll
                for (int r = 0; r < 8; ++r) sacc[t][r] = 0.0f;
            }
#pragma unroll
            for (int t = 0; t < 4; ++t) {
                const _Float16* krow = kt + (t * 16 + lm) * KST + 16 * g;
#pragma unroll
                for (int c = 0; c < 4; ++c) {
                    v16h a = load_frag16(krow + 32 * c);
                    sacc[t] = __builtin_amdgcn_wmma_f32_16x16x32_f16(
                        false, a, false, qf[c], (short)0, sacc[t], false, false);
                }
            }

            // ---- causal mask (C layout: lane holds col m=q0+lm, VGPR r -> n) ----
            const int mrow = q0 + lm;
            if (kt0 + BN - 1 > q0) {
#pragma unroll
                for (int t = 0; t < 4; ++t) {
#pragma unroll
                    for (int r = 0; r < 8; ++r) {
                        const int n = kt0 + t * 16 + 8 * g + r;
                        if (n > mrow) sacc[t][r] = -__builtin_inff();
                    }
                }
            }

            // ---- online softmax over n: regs + xor-16 partner reduction ----
            float tmax = -__builtin_inff();
#pragma unroll
            for (int t = 0; t < 4; ++t) {
#pragma unroll
                for (int r = 0; r < 8; ++r) tmax = fmaxf(tmax, sacc[t][r]);
            }
            tmax = fmaxf(tmax, __shfl_xor(tmax, 16, 32));
            const float mnew  = fmaxf(m_run, tmax);
            // raw v_exp_f32: args are always <= 0, -inf -> 0. No range fixup.
            const float alpha = __builtin_amdgcn_exp2f((m_run - mnew) * CEXP);
            const float nmc   = -mnew * CEXP;
            float lsum = 0.0f;
#pragma unroll
            for (int t = 0; t < 4; ++t) {
#pragma unroll
                for (int r = 0; r < 8; ++r) {
                    const float pe =
                        __builtin_amdgcn_exp2f(fmaf(sacc[t][r], CEXP, nmc));
                    sacc[t][r] = pe;
                    lsum += pe;
                }
            }
            lsum += __shfl_xor(lsum, 16, 32);
            l_run = l_run * alpha + lsum;
            m_run = mnew;
#pragma unroll
            for (int dt = 0; dt < 8; ++dt) {
#pragma unroll
                for (int r = 0; r < 8; ++r) oacc[dt][r] *= alpha;
            }

            // ---- O^T += V^T x P^T. P^T B-fragments built in registers. ----
#pragma unroll
            for (int c = 0; c < 2; ++c) {
                // Pack probabilities to f16x2 first, shuffle u32 pairs (4
                // ds_bpermute instead of 8 f32 shuffles).
                uint32_t Apk[4], Bpk[4];
#pragma unroll
                for (int rp = 0; rp < 4; ++rp) {
                    Apk[rp] = pk_f16x2(sacc[2 * c + 0][2 * rp], sacc[2 * c + 0][2 * rp + 1]);
                    Bpk[rp] = pk_f16x2(sacc[2 * c + 1][2 * rp], sacc[2 * c + 1][2 * rp + 1]);
                }
                union { v16h f; uint32_t u[8]; } pt;
#pragma unroll
                for (int rp = 0; rp < 4; ++rp) {
                    // our lane needs tile 2c+g; partner (xor16) holds its other 8 rows
                    const uint32_t mine = g ? Bpk[rp] : Apk[rp];
                    const uint32_t send = g ? Apk[rp] : Bpk[rp];
                    const uint32_t got  = __shfl_xor(send, 16, 32);
                    pt.u[rp]     = g ? got  : mine;  // k-halves 0..7  (group-0 rows)
                    pt.u[4 + rp] = g ? mine : got;   // k-halves 8..15 (group-1 rows)
                }
#pragma unroll
                for (int dt = 0; dt < 8; ++dt) {
                    const _Float16* vrow = vt + (dt * 16 + lm) * VST + 32 * c + 16 * g;
                    v16h a = load_frag16(vrow);
                    oacc[dt] = __builtin_amdgcn_wmma_f32_16x16x32_f16(
                        false, a, false, pt.f, (short)0, oacc[dt], false, false);
                }
            }
        }
        buf ^= 1;
    }

    __syncthreads();  // before reusing LDS for output staging

    // ---- epilogue: normalize, transpose O^T through LDS, coalesced stores ----
    const float inv = 1.0f / l_run;
    float* ostg = (float*)(smem) + wave * (16 * OST);
#pragma unroll
    for (int dt = 0; dt < 8; ++dt) {
#pragma unroll
        for (int r = 0; r < 8; ++r) {
            const int d = dt * 16 + 8 * g + r;
            ostg[lm * OST + d] = oacc[dt][r] * inv;
        }
    }
    // same-wave LDS WAR/RAW handled by DScnt waits the compiler inserts
    for (int rr = 0; rr < 16; ++rr) {
        float4 x = *(const float4*)(ostg + rr * OST + lane * 4);
        *(float4*)(O + (((size_t)b * FA_SQ + q0 + rr) * FA_H + h) * FA_D + lane * 4) = x;
    }
}

extern "C" void kernel_launch(void* const* d_in, const int* in_sizes, int n_in,
                              void* d_out, int out_size, void* d_ws, size_t ws_size,
                              hipStream_t stream) {
    const float* Q = (const float*)d_in[0];
    const float* K = (const float*)d_in[1];
    const float* V = (const float*)d_in[2];
    float* O = (float*)d_out;
    (void)in_sizes; (void)n_in; (void)out_size; (void)d_ws; (void)ws_size;

    dim3 grid(FA_SQ / BM, FA_H, FA_B);
    fa_fwd_wmma<<<grid, 256, SM_TOTAL, stream>>>(Q, K, V, O);
}
